// PositionOnlyRouter_51934744543481
// MI455X (gfx1250) — matrix operand order repaired and checked by reading, not compiled, not run on hardware
//
#include <hip/hip_runtime.h>
#include <cmath>

#define SEQ   2048
#define NEXP  8
#define DPOS  32
#define DIM   1024
#define BATCH 8
#define BLK_M 64
#define BLK_N 128
#define MAX_MB 264   // worst case: ceil((16384 + 8*(64-8))/64) = 263

typedef __attribute__((ext_vector_type(16))) __bf16 v16bf;
typedef __attribute__((ext_vector_type(8)))  float  v8f;

union Frag16 {
    v16bf v;
    unsigned short u[16];
    uint4 q[2];
};

__device__ __forceinline__ unsigned short f2bf(float f) {
    unsigned int u = __float_as_uint(f);
    u += 0x7FFFu + ((u >> 16) & 1u);     // round-to-nearest-even to bf16
    return (unsigned short)(u >> 16);
}
__device__ __forceinline__ float bf2f(unsigned short h) {
    return __uint_as_float(((unsigned int)h) << 16);
}

// ---------------------------------------------------------------------------
// Pre-pack: split fp32 into bf16 hi/lo planes (bf16x3 decomposition), hoisted
// out of the GEMM so the inner loop has zero conversion VALU. Pure streaming.
// ---------------------------------------------------------------------------
__global__ __launch_bounds__(256) void pack_kernel(
    const float* __restrict__ src,
    unsigned short* __restrict__ hi, unsigned short* __restrict__ lo,
    int n)   // n multiple of 4
{
    int i = (blockIdx.x * 256 + threadIdx.x) * 4;
    if (i >= n) return;
    const float4 v = *(const float4*)(src + i);
    float f[4] = {v.x, v.y, v.z, v.w};
    unsigned int hp[2], lp[2];
    #pragma unroll
    for (int j = 0; j < 2; ++j) {
        unsigned short h0 = f2bf(f[2*j]),   h1 = f2bf(f[2*j+1]);
        unsigned short l0 = f2bf(f[2*j]   - bf2f(h0));
        unsigned short l1 = f2bf(f[2*j+1] - bf2f(h1));
        hp[j] = (unsigned int)h0 | ((unsigned int)h1 << 16);
        lp[j] = (unsigned int)l0 | ((unsigned int)l1 << 16);
    }
    *(uint2*)(hi + i) = make_uint2(hp[0], hp[1]);
    *(uint2*)(lo + i) = make_uint2(lp[0], lp[1]);
}

// ---------------------------------------------------------------------------
// Router: per-position argmax over pe[s,:]·sign(tile_sigs), bucket the 16384
// (b,s) rows by expert into 64-row-aligned segments with -1 padding.
// ---------------------------------------------------------------------------
__global__ __launch_bounds__(256) void router_kernel(
    const float* __restrict__ tile_sigs,
    int* __restrict__ nTiles, int* __restrict__ tileExpert,
    int* __restrict__ perm)
{
    __shared__ float sgn[NEXP * DPOS];
    __shared__ int cnt[NEXP], segStart[NEXP], segEndPad[NEXP], cursor[NEXP];
    __shared__ int totBlk;
    const int tid = threadIdx.x;

    for (int i = tid; i < NEXP * DPOS; i += 256) {
        float v = tile_sigs[i];
        sgn[i] = (v > 0.f) ? 1.f : ((v < 0.f) ? -1.f : 0.f);
    }
    if (tid < NEXP) { cnt[tid] = 0; cursor[tid] = 0; }
    __syncthreads();

    unsigned char myE[SEQ / 256];
    const float kFreq = -0.28782313662425573f;   // -ln(10000)/32
    int it = 0;
    #pragma unroll
    for (int s = tid; s < SEQ; s += 256, ++it) {
        float sc[DPOS / 2], cc[DPOS / 2];
        #pragma unroll
        for (int i = 0; i < DPOS / 2; ++i) {
            float a = (float)s * __expf((float)(2 * i) * kFreq);
            sc[i] = __sinf(a); cc[i] = __cosf(a);
        }
        float best = -1e30f; int bi = 0;
        for (int t = 0; t < NEXP; ++t) {
            float sum = 0.f;
            #pragma unroll
            for (int i = 0; i < DPOS / 2; ++i)
                sum += sc[i] * sgn[t * DPOS + 2 * i] + cc[i] * sgn[t * DPOS + 2 * i + 1];
            if (sum > best) { best = sum; bi = t; }   // first-max tie break, like argmax
        }
        myE[it] = (unsigned char)bi;
        atomicAdd(&cnt[bi], 1);
    }
    __syncthreads();

    if (tid == 0) {
        int acc = 0;
        for (int t = 0; t < NEXP; ++t) {
            segStart[t] = acc;
            int rows = cnt[t] * BATCH;
            int pad  = (rows + (BLK_M - 1)) & ~(BLK_M - 1);
            segEndPad[t] = acc + pad;
            acc += pad;
        }
        totBlk = acc / BLK_M;
        *nTiles = totBlk;
    }
    __syncthreads();

    for (int blk = tid; blk < totBlk; blk += 256) {
        int row = blk * BLK_M, e = 0;
        for (int t = 0; t < NEXP; ++t)
            if (row >= segStart[t] && row < segEndPad[t]) e = t;
        tileExpert[blk] = e;
    }
    for (int t = 0; t < NEXP; ++t) {
        int from = segStart[t] + cnt[t] * BATCH;
        for (int i = from + tid; i < segEndPad[t]; i += 256) perm[i] = -1;
    }
    __syncthreads();

    it = 0;
    #pragma unroll
    for (int s = tid; s < SEQ; s += 256, ++it) {
        int e = myE[it];
        int j = atomicAdd(&cursor[e], 1);
        int base = segStart[e] + j * BATCH;
        #pragma unroll
        for (int b = 0; b < BATCH; ++b) perm[base + b] = b * SEQ + s;
    }
}

// ---------------------------------------------------------------------------
// Grouped expert GEMM on pre-split bf16 hi/lo planes.
// Block = 64 rows (single expert) x 128 cols; 8 waves, wave owns a 16-col
// strip with 4 M-subtiles. Double-buffered LDS A staging: one barrier per
// K-step, next chunk's global loads overlap the 12 WMMAs.
// Padding rows (perm == -1) load row 0 (results masked at the store).
// ---------------------------------------------------------------------------
__global__ __launch_bounds__(256) void moe_gemm_kernel(
    const unsigned short* __restrict__ xHi, const unsigned short* __restrict__ xLo,
    const unsigned short* __restrict__ wHi, const unsigned short* __restrict__ wLo,
    const float* __restrict__ bias, float* __restrict__ out,
    const int* __restrict__ nTiles, const int* __restrict__ tileExpert,
    const int* __restrict__ perm)
{
    const int nb = *nTiles;
    const int mb = blockIdx.x;
    if (mb >= nb) return;                 // uniform: EXEC stays all-ones below
    const int e     = tileExpert[mb];
    const int nbase = blockIdx.y * BLK_N;
    const int tid   = threadIdx.x;
    const int lane  = tid & 31;
    const int w     = tid >> 5;

    __shared__ __align__(16) unsigned int aHi[2][BLK_M * 16];  // 64 x 32 halves x2
    __shared__ __align__(16) unsigned int aLo[2][BLK_M * 16];
    __shared__ int   rowIdx[BLK_M];
    __shared__ float biasS[BLK_N];
    __shared__ float cTile[BLK_M * BLK_N];

    if (tid < BLK_M) rowIdx[tid] = perm[mb * BLK_M + tid];
    if (tid < BLK_N) biasS[tid]  = bias[e * DIM + nbase + tid];
    __syncthreads();

    const v8f vzero = {0.f, 0.f, 0.f, 0.f, 0.f, 0.f, 0.f, 0.f};
    v8f acc[4];
    #pragma unroll
    for (int m = 0; m < 4; ++m) acc[m] = vzero;

    // A-fragment lane pattern (ISA 16-bit A 16x32 layout)
    const int off = (lane < 16) ? 0 : 8;    // half offset within staged K-chunk
    const int kB  = (lane < 16) ? 0 : 8;    // B K offset (B 32x16 layout)
    const int rA  = lane & 15;

    // A staging roles: 4 threads per row, 8 halves (16 B) per plane each.
    // Padding rows clamp to row 0 -- loads are unconditional, outputs masked.
    const int sr = tid >> 2;
    const int sc = (tid & 3) * 8;
    const int prS = rowIdx[sr];
    const size_t aRow = (size_t)((prS >= 0) ? prS : 0) * DIM;
    const unsigned short* xHiRow = xHi + aRow;
    const unsigned short* xLoRow = xLo + aRow;
    const int sIdx = sr * 16 + sc / 2;

    // B: lane reads two runs of 8 pre-split bf16 (at k0+kB and k0+kB+16) of W[e][n][.] (L2-resident)
    const size_t wOff = ((size_t)e * DIM + (size_t)(nbase + w * 16 + rA)) * DIM;
    const unsigned short* wHiRow = wHi + wOff;
    const unsigned short* wLoRow = wLo + wOff;

    // ---- prologue: stage chunk 0 into buffer 0 ----
    *(uint4*)(&aHi[0][sIdx]) = *(const uint4*)(xHiRow + sc);
    *(uint4*)(&aLo[0][sIdx]) = *(const uint4*)(xLoRow + sc);
    __syncthreads();

    for (int k0 = 0; k0 < DIM; k0 += 32) {
        const int  buf     = (k0 >> 5) & 1;
        const bool hasNext = (k0 + 32) < DIM;      // uniform

        // ---- issue next A chunk's global loads (overlap with WMMAs) ----
        uint4 nh, nl;
        if (hasNext) {
            nh = *(const uint4*)(xHiRow + k0 + 32 + sc);
            nl = *(const uint4*)(xLoRow + k0 + 32 + sc);
        }

        // ---- B fragments: pre-split, 4 x b128 global loads ----
        Frag16 bHi, bLo;
        bHi.q[0] = *(const uint4*)(wHiRow + k0 + kB);
        bHi.q[1] = *(const uint4*)(wHiRow + k0 + kB + 16);
        bLo.q[0] = *(const uint4*)(wLoRow + k0 + kB);
        bLo.q[1] = *(const uint4*)(wLoRow + k0 + kB + 16);
        if (hasNext) {
            __builtin_prefetch(wHiRow + k0 + 32, 0, 1);
            __builtin_prefetch(wLoRow + k0 + 32, 0, 1);
        }

        // ---- 4 M-subtiles x bf16x3 WMMA on current buffer ----
        #pragma unroll
        for (int m = 0; m < 4; ++m) {
            Frag16 ah, al;
            const unsigned int* bh = &aHi[buf][(m * 16 + rA) * 16];
            const unsigned int* bl = &aLo[buf][(m * 16 + rA) * 16];
            ah.q[0] = *(const uint4*)(bh + off / 2);
            ah.q[1] = *(const uint4*)(bh + 8 + off / 2);
            al.q[0] = *(const uint4*)(bl + off / 2);
            al.q[1] = *(const uint4*)(bl + 8 + off / 2);
            acc[m] = __builtin_amdgcn_wmma_f32_16x16x32_bf16(
                false, ah.v, false, bHi.v, (short)0, acc[m], false, false);
            acc[m] = __builtin_amdgcn_wmma_f32_16x16x32_bf16(
                false, ah.v, false, bLo.v, (short)0, acc[m], false, false);
            acc[m] = __builtin_amdgcn_wmma_f32_16x16x32_bf16(
                false, al.v, false, bHi.v, (short)0, acc[m], false, false);
        }

        // ---- store next chunk into the other buffer, single barrier ----
        if (hasNext) {
            *(uint4*)(&aHi[buf ^ 1][sIdx]) = nh;
            *(uint4*)(&aLo[buf ^ 1][sIdx]) = nl;
        }
        __syncthreads();
    }

    // ---- epilogue: re-tile through LDS for coalesced stores ----
    {
        const int cl = w * 16 + (lane & 15);
        const int rb = (lane < 16) ? 0 : 8;      // C/D layout: vgpr j -> M=j / M=8+j
        #pragma unroll
        for (int m = 0; m < 4; ++m)
            #pragma unroll
            for (int j = 0; j < 8; ++j)
                cTile[(m * 16 + rb + j) * BLK_N + cl] = acc[m][j];
    }
    __syncthreads();
    for (int idx = tid; idx < BLK_M * BLK_N; idx += 256) {
        int r = idx >> 7, c = idx & (BLK_N - 1);
        int pr = rowIdx[r];
        if (pr >= 0)
            out[(size_t)pr * DIM + nbase + c] = cTile[idx] + biasS[c];
    }
}

extern "C" void kernel_launch(void* const* d_in, const int* in_sizes, int n_in,
                              void* d_out, int out_size, void* d_ws, size_t ws_size,
                              hipStream_t stream) {
    (void)in_sizes; (void)n_in; (void)out_size; (void)ws_size;
    const float* x  = (const float*)d_in[0];
    const float* ts = (const float*)d_in[1];
    const float* W  = (const float*)d_in[2];
    const float* b  = (const float*)d_in[3];
    float* out = (float*)d_out;

    // ---- workspace layout ----
    const int NW = NEXP * DIM * DIM;          // 8M  W elements
    const int NX = BATCH * SEQ * DIM;         // 16M x elements
    char* base = (char*)d_ws;
    int* wsI        = (int*)base;
    int* nTiles     = wsI;
    int* tileExpert = wsI + 1;
    int* perm       = wsI + 1 + MAX_MB;       // 16896 ints
    size_t planeOff = 128 * 1024;             // int region < 80 KB, pad to 128 KB
    unsigned short* wHiP = (unsigned short*)(base + planeOff);
    unsigned short* wLoP = wHiP + NW;
    unsigned short* xHiP = wLoP + NW;
    unsigned short* xLoP = xHiP + NX;         // total ~96.1 MB

    // ---- pre-pack fp32 -> bf16 hi/lo planes (streaming, once per launch) ----
    pack_kernel<<<NW / 1024, 256, 0, stream>>>(W, wHiP, wLoP, NW);
    pack_kernel<<<NX / 1024, 256, 0, stream>>>(x, xHiP, xLoP, NX);

    // ---- routing ----
    router_kernel<<<1, 256, 0, stream>>>(ts, nTiles, tileExpert, perm);

    // ---- grouped expert GEMM ----
    dim3 grid(MAX_MB, DIM / BLK_N);           // extra M-blocks exit on nTiles check
    moe_gemm_kernel<<<grid, 256, 0, stream>>>(xHiP, xLoP, wHiP, wLoP,
                                              b, out, nTiles, tileExpert, perm);
}
